// GraphConvLayer_51264729645566
// MI455X (gfx1250) — matrix operand order repaired and checked by reading, not compiled, not run on hardware
//
#include <hip/hip_runtime.h>
#include <hip/hip_bf16.h>

// ---- CDNA5 WMMA types ----
typedef __attribute__((ext_vector_type(16))) __bf16 v16bf;
typedef __attribute__((ext_vector_type(8)))  float  v8f;

__device__ __forceinline__ __bf16 f2bf(float f) { return (__bf16)f; }

__device__ __forceinline__ v8f wmma_bf16(v16bf a, v16bf b, v8f c) {
  // D = A(16x32 bf16) x B(32x16 bf16) + C(16x16 f32)
  return __builtin_amdgcn_wmma_f32_16x16x32_bf16(
      /*neg_a=*/false, a, /*neg_b=*/false, b,
      /*c_mod=*/(short)0, c, /*reuse_a=*/false, /*reuse_b=*/false);
}

// =====================================================================
// Generic row-major fp32->bf16 WMMA GEMM:
//   C[M,N] = op(A[M,K] @ B[K,N] + bias[col] * (HAS_RS ? rowscale[row] : 1))
// Compile-time epilogue flags -> branch-free epilogue (no per-element
// null-checks / scalar branches).
// One wave computes one 16x16 output tile. 4 waves / block.
// =====================================================================
template<bool HAS_RS, bool RELU>
__global__ void gemm_bf16_wmma_kernel(const float* __restrict__ A, int lda,
                                      const float* __restrict__ Bm, int ldb,
                                      float* __restrict__ C, int ldc,
                                      int M, int N, int K,
                                      const float* __restrict__ bias,
                                      const float* __restrict__ rowscale)
{
  const int lane = threadIdx.x & 31;
  const int wave = threadIdx.x >> 5;
  const int ntn  = N >> 4;
  const int total = (M >> 4) * ntn;
  const int tile  = blockIdx.x * 4 + wave;   // wave-uniform
  if (tile >= total) return;
  const int tm = tile / ntn;
  const int tn = tile - tm * ntn;

  const int half = lane >> 4;          // 0 | 1
  const int l16  = lane & 15;
  const int arow = tm * 16 + l16;      // A row held by this lane
  const int bcol = tn * 16 + l16;      // B/C column held by this lane
  const int a_kb = half * 16;          // A K-base within 32-chunk: {0,16}, 16 consecutive K per lane
  const int b_kb = half * 16;          // B K-base: {0,16}, 16 consecutive K per lane

  v8f acc = {};
  for (int k0 = 0; k0 < K; k0 += 32) {
    // A fragment (16x32 bf16): two contiguous 8-float runs per lane
    v16bf af;
    const float* ap = A + (size_t)arow * lda + k0;
#pragma unroll
    for (int e = 0; e < 8; ++e) af[e]     = f2bf(ap[a_kb + e]);
#pragma unroll
    for (int e = 0; e < 8; ++e) af[8 + e] = f2bf(ap[8 + a_kb + e]);
    // B fragment (32x16 bf16): 16 K-strided loads of this lane's column
    v16bf bfr;
    const float* bp = Bm + (size_t)(k0 + b_kb) * ldb + bcol;
#pragma unroll
    for (int e = 0; e < 16; ++e) bfr[e] = f2bf(bp[(size_t)e * ldb]);

    acc = wmma_bf16(af, bfr, acc);
  }

  // ---- branch-free epilogue ----
  const float bc = bias[bcol];
  const int row0 = tm * 16 + half * 8;     // this lane's first C row
  float rs[8];
  if (HAS_RS) {
#pragma unroll
    for (int v = 0; v < 8; ++v) rs[v] = rowscale[row0 + v];
  }
#pragma unroll
  for (int v = 0; v < 8; ++v) {
    float val = acc[v] + (HAS_RS ? bc * rs[v] : bc);
    if (RELU) val = fmaxf(val, 0.0f);
    C[(size_t)(row0 + v) * ldc + bcol] = val;
  }
}

// =====================================================================
// Message + aggregation kernel. One workgroup (256 thr = 8 waves) per (b,i).
//   H[j,h] = relu( P[bi,h] + edge[bi,j,:] @ W1e[:,h] )      (K = E = 32)
//   G[bi,h] = sum_j adj[bi,j] * H[j,h] ;  RS[bi] = sum_j adj[bi,j]
// Edge slice (256x32) and W1e (32x256) staged in LDS as bf16.
// Each wave owns 2 h-tiles (32 h cols), loops 16 j-tiles -> 32 WMMAs/wave.
// =====================================================================
__global__ void msg_kernel(const float* __restrict__ edge,  // [B,N,N,E]
                           const float* __restrict__ adj,   // [B,N,N]
                           const float* __restrict__ P,     // [B*N,256] (incl. b1)
                           const float* __restrict__ W1,    // [160,256]
                           float* __restrict__ G,           // [B*N,256]
                           float* __restrict__ RS)          // [B*N]
{
  __shared__ __align__(16) __bf16 Ae[256 * 32];   // edge slice, row j, 32 k — 16KB
  __shared__ __align__(16) __bf16 We[32 * 256];   // W1 rows 128..159      — 16KB
  __shared__ float a_row[256];
  __shared__ float p_row[256];

  const int bi   = blockIdx.x;          // b*N + i
  const int tid  = threadIdx.x;
  const int lane = tid & 31;
  const int wave = tid >> 5;
  const int half = lane >> 4;
  const int l16  = lane & 15;

  // ---- stage inputs to LDS (coalesced float4, cvt to bf16) ----
  const float* esrc = edge + (size_t)bi * (256 * 32);
  for (int f = tid * 4; f < 256 * 32; f += 256 * 4) {
    float4 v = *(const float4*)(esrc + f);
    Ae[f + 0] = f2bf(v.x); Ae[f + 1] = f2bf(v.y);
    Ae[f + 2] = f2bf(v.z); Ae[f + 3] = f2bf(v.w);
  }
  const float* wsrc = W1 + 128 * 256;
  for (int f = tid * 4; f < 32 * 256; f += 256 * 4) {
    float4 v = *(const float4*)(wsrc + f);
    We[f + 0] = f2bf(v.x); We[f + 1] = f2bf(v.y);
    We[f + 2] = f2bf(v.z); We[f + 3] = f2bf(v.w);
  }
  a_row[tid] = adj[(size_t)bi * 256 + tid];
  p_row[tid] = P[(size_t)bi * 256 + tid];
  __syncthreads();

  // ---- adjacency row sum (wave 0) ----
  if (wave == 0) {
    float s = 0.0f;
#pragma unroll
    for (int t = lane; t < 256; t += 32) s += a_row[t];
#pragma unroll
    for (int off = 16; off > 0; off >>= 1) s += __shfl_down(s, off, 32);
    if (lane == 0) RS[bi] = s;
  }

  // ---- this wave's two h-tiles ----
  const int ht0 = wave * 32;
  const int ht1 = wave * 32 + 16;
  const int n0 = ht0 + l16;
  const int n1 = ht1 + l16;
  const int b_kb = half * 16;
  const int a_kb = half * 16;

  // B fragments from LDS (invariant over j-tiles)
  v16bf bf0, bf1;
#pragma unroll
  for (int e = 0; e < 16; ++e) {
    bf0[e] = We[(b_kb + e) * 256 + n0];
    bf1[e] = We[(b_kb + e) * 256 + n1];
  }
  const float p0 = p_row[n0];
  const float p1 = p_row[n1];

  float g0 = 0.0f, g1 = 0.0f;
#pragma unroll 1
  for (int jt = 0; jt < 16; ++jt) {
    // A fragment: row j = jt*16 + l16 of edge slice
    v16bf af;
    const __bf16* ap = &Ae[(jt * 16 + l16) * 32];
#pragma unroll
    for (int e = 0; e < 8; ++e) af[e]     = ap[a_kb + e];
#pragma unroll
    for (int e = 0; e < 8; ++e) af[8 + e] = ap[8 + a_kb + e];

    v8f z = {};
    v8f d0 = wmma_bf16(af, bf0, z);
    v8f d1 = wmma_bf16(af, bf1, z);

    // relu + adjacency-weighted reduction over the 8 j-rows this lane holds
#pragma unroll
    for (int v = 0; v < 8; ++v) {
      const float av = a_row[jt * 16 + v + half * 8];
      g0 += av * fmaxf(d0[v] + p0, 0.0f);
      g1 += av * fmaxf(d1[v] + p1, 0.0f);
    }
  }

  // combine the two row-halves (lanes L and L+16 hold the same column)
  g0 += __shfl_xor(g0, 16, 32);
  g1 += __shfl_xor(g1, 16, 32);
  if (lane < 16) {
    G[(size_t)bi * 256 + ht0 + lane] = g0;
    G[(size_t)bi * 256 + ht1 + lane] = g1;
  }
}

// Copy node_features [1024,128] into UpdIn[:,0:128] (ldc=256)
__global__ void copy_node_kernel(const float* __restrict__ node,
                                 float* __restrict__ updin)
{
  const int idx = blockIdx.x * blockDim.x + threadIdx.x; // 32768 float4's
  const int f = idx * 4;
  const int row = f >> 7;        // /128
  const int col = f & 127;
  float4 v = *(const float4*)(node + f);
  *(float4*)(updin + (size_t)row * 256 + col) = v;
}

extern "C" void kernel_launch(void* const* d_in, const int* in_sizes, int n_in,
                              void* d_out, int out_size, void* d_ws, size_t ws_size,
                              hipStream_t stream)
{
  const float* node = (const float*)d_in[0];   // [4,256,128]
  const float* adj  = (const float*)d_in[1];   // [4,256,256]
  const float* edge = (const float*)d_in[2];   // [4,256,256,32]
  const float* W1   = (const float*)d_in[3];   // [160,256]
  const float* b1   = (const float*)d_in[4];   // [256]
  const float* W2   = (const float*)d_in[5];   // [256,128]
  const float* b2   = (const float*)d_in[6];   // [128]
  const float* U1   = (const float*)d_in[7];   // [256,256]
  const float* bu1  = (const float*)d_in[8];   // [256]
  const float* U2   = (const float*)d_in[9];   // [256,128]
  const float* bu2  = (const float*)d_in[10];  // [128]
  float* out = (float*)d_out;                  // [4,256,128] = [1024,128]

  float* ws    = (float*)d_ws;
  float* P     = ws;                    // [1024,256]
  float* G     = P     + 1024 * 256;    // [1024,256]
  float* UpdIn = G     + 1024 * 256;    // [1024,256]
  float* Uu    = UpdIn + 1024 * 256;    // [1024,256]
  float* RS    = Uu    + 1024 * 256;    // [1024]

  // 1) P = node @ W1[0:128,:] + b1           (M=1024,N=256,K=128)
  gemm_bf16_wmma_kernel<false, false><<<(64 * 16) / 4, 128, 0, stream>>>(
      node, 128, W1, 256, P, 256, 1024, 256, 128, b1, nullptr);

  // 2) UpdIn[:,0:128] = node
  copy_node_kernel<<<128, 256, 0, stream>>>(node, UpdIn);

  // 3) G[bi,h] = sum_j adj*relu(P + edge@W1e) ; RS[bi] = sum_j adj
  msg_kernel<<<1024, 256, 0, stream>>>(edge, adj, P, W1, G, RS);

  // 4) UpdIn[:,128:256] = G @ W2 + RS*b2     (M=1024,N=128,K=256)
  gemm_bf16_wmma_kernel<true, false><<<(64 * 8) / 4, 128, 0, stream>>>(
      G, 256, W2, 128, UpdIn + 128, 256, 1024, 128, 256, b2, RS);

  // 5) Uu = relu(UpdIn @ U1 + bu1)           (M=1024,N=256,K=256)
  gemm_bf16_wmma_kernel<false, true><<<(64 * 16) / 4, 128, 0, stream>>>(
      UpdIn, 256, U1, 256, Uu, 256, 1024, 256, 256, bu1, nullptr);

  // 6) out = Uu @ U2 + bu2                   (M=1024,N=128,K=256)
  gemm_bf16_wmma_kernel<false, false><<<(64 * 8) / 4, 128, 0, stream>>>(
      Uu, 256, U2, 128, out, 128, 1024, 128, 256, bu2, nullptr);
}
